// Net_view_83623013253546
// MI455X (gfx1250) — matrix operand order repaired and checked by reading, not compiled, not run on hardware
//
#include <hip/hip_runtime.h>
#include <hip/hip_bf16.h>

// ---------------------------------------------------------------------------
// MI455X (gfx1250) implementation: implicit-GEMM convolutions on
// v_wmma_f32_16x16x32_f16 (wave32), NHWC-f16 activations with channel pad
// (mult of 32) and 2-pixel zero halo (52x52) for branch-free B128 im2col
// loads. Weights pre-packed as A fragments in contiguous k order (k = 16*hi + e).
// ---------------------------------------------------------------------------

typedef _Float16 v16h __attribute__((ext_vector_type(16)));
typedef _Float16 v8h  __attribute__((ext_vector_type(8)));
typedef float    v8f  __attribute__((ext_vector_type(8)));

#define HPAD 52               // 48 + 2*2 halo
#define PIX  (HPAD*HPAD)      // 2704 padded pixels per image

// ---------------------------------------------------------------- bilinear --
__device__ __forceinline__ float bilin48(const float* __restrict__ img,
                                         float ix, float iy) {
  float x0f = floorf(ix), y0f = floorf(iy);
  float wx = ix - x0f,    wy = iy - y0f;
  int x0 = (int)x0f, y0 = (int)y0f;
  int x1 = x0 + 1,   y1 = y0 + 1;
  float v00 = (x0 >= 0 && x0 < 48 && y0 >= 0 && y0 < 48) ? img[y0*48 + x0] : 0.f;
  float v01 = (x1 >= 0 && x1 < 48 && y0 >= 0 && y0 < 48) ? img[y0*48 + x1] : 0.f;
  float v10 = (x0 >= 0 && x0 < 48 && y1 >= 0 && y1 < 48) ? img[y1*48 + x0] : 0.f;
  float v11 = (x1 >= 0 && x1 < 48 && y1 >= 0 && y1 < 48) ? img[y1*48 + x1] : 0.f;
  return (1.f-wy)*((1.f-wx)*v00 + wx*v01) + wy*((1.f-wx)*v10 + wx*v11);
}

// ------------------------------------------------------ PSV (plane sweep) --
// Writes chunk of the 200-image PSV as NHWC-f16 (Cpad=32, ch0..3 valid) into
// a pre-zeroed padded buffer.
__global__ void psv_kernel(const float* __restrict__ img, const int* __restrict__ ind,
                           _Float16* __restrict__ out, int b0, int G) {
  int idx = blockIdx.x * blockDim.x + threadIdx.x;
  if (idx >= G * 2304) return;
  int bc = idx / 2304, p = idx % 2304;
  int y = p / 48, x = p % 48;
  int b = b0 + bc;
  int a = b >> 3, d = b & 7;                 // batch = a*8 + d
  float disp = -4.f + (float)d * (8.f / 7.f);
  float t_w = (float)(a % 5), t_h = (float)(a / 5);
  long obase = (((long)bc * HPAD + (y + 2)) * HPAD + (x + 2)) * 32;
  #pragma unroll
  for (int s = 0; s < 4; ++s) {
    float sv  = (float)ind[s];
    float s_h = floorf(sv / 5.f);
    float s_w = sv - 5.f * s_h;
    float offw = disp * (t_w - s_w);
    float offh = disp * (t_h - s_h);
    float gx = 2.f * ((float)x + offw) / 47.f - 1.f;
    float gy = 2.f * ((float)y + offh) / 47.f - 1.f;
    float ix = ((gx + 1.f) * 48.f - 1.f) * 0.5f;
    float iy = ((gy + 1.f) * 48.f - 1.f) * 0.5f;
    out[obase + s] = (_Float16)bilin48(img + s * 2304, ix, iy);
  }
}

// -------------------------------------------------------- weight packing ---
// dst[((mt*KH*KW + kpos)*KB + kb)*512 + lane*16 + e] holds
// W[co = mt*16 + (lane&15)][ci = kb*32 + 16*(lane>>4) + e][ky][kx]
// i.e. contiguous k order for the 16x32 A operand, same as the B loads (zero pad OOB).
__global__ void pack_w_kernel(const float* __restrict__ w, _Float16* __restrict__ dst,
                              int Cout, int Cin, int KH, int KW, int KB,
                              int mtiles, int total) {
  int idx = blockIdx.x * blockDim.x + threadIdx.x;
  if (idx >= total) return;
  int e    = idx & 15;
  int lane = (idx >> 4) & 31;
  int rest = idx >> 9;
  int kb   = rest % KB;      rest /= KB;
  int kpos = rest % (KH*KW); int mt = rest / (KH*KW);
  int ky = kpos / KW, kx = kpos % KW;
  int hi = lane >> 4, l15 = lane & 15;
  int co = mt * 16 + l15;
  int ci = kb * 32 + 16 * hi + e;
  float v = 0.f;
  if (co < Cout && ci < Cin) v = w[((co * Cin + ci) * KH + ky) * KW + kx];
  dst[idx] = (_Float16)v;
}

// ----------------------------------------------------- WMMA implicit GEMM --
// One wave computes a 16(M) x 64(N) output tile. N = 2304 pixels = 36 tiles
// of 64. K loops over (ky,kx) x 32-channel blocks; per step: 1 A-frag load
// (packed weights, contiguous 32B) + 4 B-frag loads (NHWC shifted pixel,
// contiguous 32B) + 4 v_wmma_f32_16x16x32_f16.
__global__ __launch_bounds__(256)
void conv_wmma_kernel(const _Float16* __restrict__ in, const _Float16* __restrict__ wp,
                      const float* __restrict__ bias,
                      _Float16* __restrict__ outh, float* __restrict__ outf,
                      int store_f32, int B, int mtiles, int KH, int KW, int pad,
                      int KB, int CpadIn, int CpadOut, int Cout, int relu) {
  const int lane = threadIdx.x & 31;
  const int hi   = lane >> 4;
  const int l15  = lane & 15;
  const int wave = blockIdx.x * (blockDim.x >> 5) + (threadIdx.x >> 5);
  const int units = B * mtiles * 36;
  if (wave >= units) return;                 // wave-uniform; EXEC stays full

  int b   = wave / (mtiles * 36);
  int rem = wave % (mtiles * 36);
  int mt  = rem / 36;
  int nt  = rem % 36;
  int nbase = nt * 64;
  const int ofs = 2 - pad;                   // halo offset correction

  int  ys[4], xs[4];
  long bases[4];
  #pragma unroll
  for (int s = 0; s < 4; ++s) {
    int n = nbase + s * 16 + l15;
    int y = n / 48, x = n % 48;
    ys[s] = y; xs[s] = x;
    bases[s] = ((long)b * HPAD + (y + ofs)) * HPAD + (x + ofs);
  }

  v8f acc[4] = {};
  const long wmt = (long)mt * (KH * KW) * KB * 512 + lane * 16;

  for (int ky = 0; ky < KH; ++ky) {
    for (int kx = 0; kx < KW; ++kx) {
      int  kpos  = ky * KW + kx;
      long delta = (long)ky * HPAD + kx;
      for (int kb = 0; kb < KB; ++kb) {
        v16h a = *(const v16h*)(wp + wmt + ((long)kpos * KB + kb) * 512);
        long choff = (long)kb * 32 + 16 * hi;
        #pragma unroll
        for (int s = 0; s < 4; ++s) {
          v16h bf = *(const v16h*)(in + (bases[s] + delta) * CpadIn + choff);
          acc[s] = __builtin_amdgcn_wmma_f32_16x16x32_f16(
              false, a, false, bf, (short)0, acc[s], false, false);
        }
      }
    }
  }

  const int chbase = mt * 16 + 8 * hi;       // D layout: M = r + 8*hi
  #pragma unroll
  for (int s = 0; s < 4; ++s) {
    long pix = ((long)b * HPAD + (ys[s] + 2)) * HPAD + (xs[s] + 2);
    if (!store_f32) {
      v8h hv;
      #pragma unroll
      for (int r = 0; r < 8; ++r) {
        int co = chbase + r;
        float v = 0.f;
        if (co < Cout) { v = acc[s][r] + bias[co]; if (relu) v = fmaxf(v, 0.f); }
        hv[r] = (_Float16)v;
      }
      *(v8h*)(outh + pix * CpadOut + chbase) = hv;
    } else {
      #pragma unroll
      for (int r = 0; r < 8; ++r) {
        int co = chbase + r;
        float v = 0.f;
        if (co < Cout) { v = acc[s][r] + bias[co]; if (relu) v = fmaxf(v, 0.f); }
        outf[pix * CpadOut + co] = v;
      }
    }
  }
}

// --------------------------------------------- pp4 (200,16) -> cp-in (25,32)
// view (200,4,h,w) as (25, 8*4, h, w): channel = d*4 + c, d = b%8.
__global__ void repack_kernel(const _Float16* __restrict__ x4,
                              _Float16* __restrict__ cpin) {
  int idx = blockIdx.x * blockDim.x + threadIdx.x;
  if (idx >= 25 * PIX * 32) return;
  int c32 = idx & 31;
  int pix = (idx >> 5) % PIX;
  int a   = idx / (PIX * 32);
  int d = c32 >> 2, c = c32 & 3;
  cpin[idx] = x4[((long)(a * 8 + d) * PIX + pix) * 16 + c];
}

// ------------------------------------------- softmax + warp + blend (f32) --
__global__ void final_kernel(const float* __restrict__ head,   // (25,HPAD,HPAD,16)
                             const float* __restrict__ img, const int* __restrict__ ind,
                             float* __restrict__ out) {
  int idx = blockIdx.x * blockDim.x + threadIdx.x;
  if (idx >= 25 * 2304) return;
  int a = idx / 2304, p = idx % 2304;
  int y = p / 48, x = p % 48;
  long base = (((long)a * HPAD + (y + 2)) * HPAD + (x + 2)) * 16;
  float disp = head[base + 0];
  float l[4], es[4];
  float m = -1e30f;
  #pragma unroll
  for (int s = 0; s < 4; ++s) { l[s] = head[base + 1 + s]; m = fmaxf(m, l[s]); }
  float sum = 0.f;
  #pragma unroll
  for (int s = 0; s < 4; ++s) { es[s] = expf(l[s] - m); sum += es[s]; }
  float t_w = (float)(a % 5), t_h = (float)(a / 5);
  float inter = 0.f;
  #pragma unroll
  for (int s = 0; s < 4; ++s) {
    float sv  = (float)ind[s];
    float s_h = floorf(sv / 5.f);
    float s_w = sv - 5.f * s_h;
    float dw = t_w - s_w, dh = t_h - s_h;
    float gx = 2.f * ((float)x + disp * dw) / 47.f - 1.f;
    float gy = 2.f * ((float)y + disp * dh) / 47.f - 1.f;
    float ix = ((gx + 1.f) * 48.f - 1.f) * 0.5f;
    float iy = ((gy + 1.f) * 48.f - 1.f) * 0.5f;
    inter += bilin48(img + s * 2304, ix, iy) * (es[s] / sum);
  }
  out[a * 2304 + p] = disp;                  // disp_target
  out[57600 + a * 2304 + p] = inter;         // inter_lf
}

// ---------------------------------------------------------------------------
extern "C" void kernel_launch(void* const* d_in, const int* in_sizes, int n_in,
                              void* d_out, int out_size, void* d_ws, size_t ws_size,
                              hipStream_t stream) {
  (void)in_sizes; (void)n_in; (void)out_size; (void)ws_size;
  const float* img = (const float*)d_in[0];
  const int*   ind = (const int*)d_in[1];
  float*       out = (float*)d_out;

  // ---- workspace carving (deterministic) ----
  size_t off = 0;
  auto carve = [&](size_t bytes) -> char* {
    char* p = (char*)d_ws + off;
    off += (bytes + 255) & ~(size_t)255;
    return p;
  };

  // layer configs: {w d_in idx, Cout, Cin, K, pad, KB, CpadIn, CpadOut, mtiles, relu}
  struct Cfg { int wi, Cout, Cin, K, pad, KB, CpadIn, CpadOut, mt, relu; };
  const Cfg cfg[11] = {
    {  2,  32,   4, 5, 2, 1,  32,  32,  2, 1 },   // pp1
    {  4,  64,  32, 5, 2, 1,  32,  64,  4, 1 },   // pp2
    {  6,  32,  64, 5, 2, 2,  64,  32,  2, 1 },   // pp3
    {  8,   4,  32, 5, 2, 1,  32,  16,  1, 1 },   // pp4
    { 10, 200,  32, 3, 1, 1,  32, 224, 13, 1 },   // cp1
    { 12, 200, 200, 3, 1, 7, 224, 224, 13, 1 },   // cp2
    { 14,   8, 200, 5, 2, 7, 224,  32,  1, 1 },   // cp3
    { 16,  64,   8, 5, 2, 1,  32,  64,  4, 1 },   // dp1
    { 18,  32,  64, 5, 2, 2,  64,  32,  2, 1 },   // dp2
    { 20,  16,  32, 3, 1, 1,  32,  32,  1, 1 },   // dp3
    { 22,   5,  16, 3, 1, 1,  32,  16,  1, 0 },   // dp4
  };

  _Float16* wp[11];
  for (int i = 0; i < 11; ++i) {
    int total = cfg[i].mt * cfg[i].K * cfg[i].K * cfg[i].KB * 512;
    wp[i] = (_Float16*)carve((size_t)total * 2);
    pack_w_kernel<<<(total + 255) / 256, 256, 0, stream>>>(
        (const float*)d_in[cfg[i].wi], wp[i],
        cfg[i].Cout, cfg[i].Cin, cfg[i].K, cfg[i].K, cfg[i].KB, cfg[i].mt, total);
  }

  _Float16* A    = (_Float16*)carve((size_t)25 * PIX * 224 * 2);   // 30.3 MB
  _Float16* Bb   = (_Float16*)carve((size_t)25 * PIX * 224 * 2);   // 30.3 MB
  _Float16* X4   = (_Float16*)carve((size_t)200 * PIX * 16 * 2);   // 17.3 MB
  _Float16* CPIN = (_Float16*)carve((size_t)25 * PIX * 32 * 2);    //  4.3 MB

  auto conv = [&](const _Float16* in, int li, void* outp, int B, int storeF32) {
    const Cfg& c = cfg[li];
    int units  = B * c.mt * 36;
    int blocks = (units + 7) / 8;
    conv_wmma_kernel<<<blocks, 256, 0, stream>>>(
        in, wp[li], (const float*)d_in[cfg[li].wi + 1],
        storeF32 ? nullptr : (_Float16*)outp,
        storeF32 ? (float*)outp : nullptr,
        storeF32, B, c.mt, c.K, c.K, c.pad, c.KB, c.CpadIn, c.CpadOut,
        c.Cout, c.relu);
  };
  auto zero = [&](void* p, size_t bytes) { hipMemsetAsync(p, 0, bytes, stream); };

  // ---- pp stage, chunked over batch (4 x 50 images) ----
  zero(X4, (size_t)200 * PIX * 16 * 2);
  const int G = 50;
  for (int ch = 0; ch < 4; ++ch) {
    int b0 = ch * G;
    zero(A, (size_t)G * PIX * 32 * 2);
    psv_kernel<<<(G * 2304 + 255) / 256, 256, 0, stream>>>(img, ind, A, b0, G);
    zero(Bb, (size_t)G * PIX * 32 * 2);
    conv(A, 0, Bb, G, 0);                                   // pp1
    zero(A, (size_t)G * PIX * 64 * 2);
    conv(Bb, 1, A, G, 0);                                   // pp2
    zero(Bb, (size_t)G * PIX * 32 * 2);
    conv(A, 2, Bb, G, 0);                                   // pp3
    conv(Bb, 3, X4 + (size_t)b0 * PIX * 16, G, 0);          // pp4 -> X4 slice
  }

  // ---- cp / dp stages (batch 25) ----
  repack_kernel<<<(25 * PIX * 32 + 255) / 256, 256, 0, stream>>>(X4, CPIN);
  zero(A, (size_t)25 * PIX * 224 * 2);
  conv(CPIN, 4, A, 25, 0);                                  // cp1
  zero(Bb, (size_t)25 * PIX * 224 * 2);
  conv(A, 5, Bb, 25, 0);                                    // cp2
  zero(A, (size_t)25 * PIX * 32 * 2);
  conv(Bb, 6, A, 25, 0);                                    // cp3
  zero(Bb, (size_t)25 * PIX * 64 * 2);
  conv(A, 7, Bb, 25, 0);                                    // dp1
  zero(A, (size_t)25 * PIX * 32 * 2);
  conv(Bb, 8, A, 25, 0);                                    // dp2
  zero(Bb, (size_t)25 * PIX * 32 * 2);
  conv(A, 9, Bb, 25, 0);                                    // dp3
  zero(A, (size_t)25 * PIX * 16 * 4);                       // dp4 head (f32)
  conv(Bb, 10, A, 25, 1);                                   // dp4 -> f32 in A

  final_kernel<<<(25 * 2304 + 255) / 256, 256, 0, stream>>>(
      (const float*)A, img, ind, out);
}
